// GATNet_65094524338520
// MI455X (gfx1250) — matrix operand order repaired and checked by reading, not compiled, not run on hardware
//
#include <hip/hip_runtime.h>
#include <hip/hip_bf16.h>
#include <math.h>

typedef __attribute__((ext_vector_type(16))) _Float16 v16h;
typedef __attribute__((ext_vector_type(8)))  float    v8f;

// ---------------- problem constants (match reference) ----------------
constexpr int NN   = 20000;           // nodes
constexpr int INC  = 512;             // layer1 in features
constexpr int HID  = 64;              // per-head hidden
constexpr int H1H  = 8;               // heads layer1
constexpr int OUTC = 256;             // layer2 out features
constexpr int NE   = 320000;          // raw edges
constexpr int EA   = NE + NN;         // edges incl. self loops
constexpr int HC1  = H1H * HID;       // 512
constexpr int MT   = NN / 16;         // 1250 M-tiles
constexpr int KT1  = INC / 32;        // 16 K-tiles gemm1
constexpr int NT1  = HC1 / 16;        // 32 N-tiles gemm1
constexpr int KT2  = HC1 / 32;        // 16 K-tiles gemm2
constexpr int NT2  = OUTC / 16;       // 16 N-tiles gemm2
constexpr float NEG = 0.2f;

// ---------------- workspace layout (all 256B aligned) ----------------
constexpr size_t SZ_XFRAG = (size_t)NN * INC * 2;     // 20,480,000 (reused as h2 frags)
constexpr size_t SZ_W1F   = (size_t)INC * HC1 * 2;    //    524,288
constexpr size_t SZ_W2F   = (size_t)HC1 * OUTC * 2;   //    262,144
constexpr size_t SZ_H1    = (size_t)NN * HC1 * 4;     // 40,960,000 (reused: g = first 20.48MB)
constexpr size_t SZ_AL    = (size_t)NN * H1H * 4;     //    640,000
constexpr size_t SZ_EDGE  = (size_t)EA * H1H * 4;     // 10,880,000
constexpr size_t SZ_AGG   = (size_t)NN * HC1 * 4;     // 40,960,000 (reused: agg2 = first 20.48MB)

constexpr size_t OFF_XFRAG = 0;
constexpr size_t OFF_W1F   = OFF_XFRAG + SZ_XFRAG;
constexpr size_t OFF_W2F   = OFF_W1F   + SZ_W1F;
constexpr size_t OFF_H1    = OFF_W2F   + SZ_W2F;
constexpr size_t OFF_AS    = OFF_H1    + SZ_H1;
constexpr size_t OFF_AD    = OFF_AS    + SZ_AL;
constexpr size_t OFF_AMAX  = OFF_AD    + SZ_AL;
constexpr size_t OFF_DEN   = OFF_AMAX  + SZ_AL;
constexpr size_t OFF_EDGE  = OFF_DEN   + SZ_AL;
constexpr size_t OFF_AGG   = OFF_EDGE  + SZ_EDGE;
// total ~116.6 MB

// ---------------- helpers ----------------
static __device__ __forceinline__ float eluf(float x) {
  return x > 0.f ? x : (expf(x) - 1.f);
}

__global__ void k_fill(float* __restrict__ p, float v, int n) {
  int i = blockIdx.x * blockDim.x + threadIdx.x;
  if (i < n) p[i] = v;
}

// Pack a row-major f32 MxK matrix into WMMA A-fragments (f16, 16x32 tiles).
// Fragment element layout (ISA 7.12.2, 16-bit A 16x32):
//   lane<16 : v[0..7]=K(0..7),   v[8..15]=K(16..23)   of row M=lane
//   lane>=16: v[0..7]=K(8..15),  v[8..15]=K(24..31)   of row M=lane-16
__global__ void k_pack_a(const float* __restrict__ src, _Float16* __restrict__ dst,
                         int Mtiles, int Ktiles, int ld) {
  int tid = blockIdx.x * blockDim.x + threadIdx.x;
  if (tid >= Mtiles * Ktiles * 32) return;
  int lane = tid & 31, tile = tid >> 5;
  int kt = tile % Ktiles, mt = tile / Ktiles;
  int r = lane & 15, hi = lane >> 4;
  const float* rowp = src + (size_t)(mt * 16 + r) * ld;
  int cb = kt * 32 + hi * 8;
  v16h v;
#pragma unroll
  for (int j = 0; j < 8; ++j) v[j]     = (_Float16)rowp[cb + j];
#pragma unroll
  for (int j = 0; j < 8; ++j) v[j + 8] = (_Float16)rowp[cb + 16 + j];
  *(v16h*)(dst + (size_t)tid * 16) = v;
}

// Same A-fragment packing, but source = agg + bias then ELU (layer1 epilogue).
__global__ void k_pack_a_elu_bias(const float* __restrict__ src, const float* __restrict__ bias,
                                  _Float16* __restrict__ dst, int Mtiles, int Ktiles, int ld) {
  int tid = blockIdx.x * blockDim.x + threadIdx.x;
  if (tid >= Mtiles * Ktiles * 32) return;
  int lane = tid & 31, tile = tid >> 5;
  int kt = tile % Ktiles, mt = tile / Ktiles;
  int r = lane & 15, hi = lane >> 4;
  const float* rowp = src + (size_t)(mt * 16 + r) * ld;
  int cb = kt * 32 + hi * 8;
  v16h v;
#pragma unroll
  for (int j = 0; j < 8; ++j) { int k = cb + j;      v[j]     = (_Float16)eluf(rowp[k] + bias[k]); }
#pragma unroll
  for (int j = 0; j < 8; ++j) { int k = cb + 16 + j; v[j + 8] = (_Float16)eluf(rowp[k] + bias[k]); }
  *(v16h*)(dst + (size_t)tid * 16) = v;
}

// Pack row-major f32 KxN weights into WMMA B-fragments (f16, 32x16 tiles).
//   lane<16 : v[0..7]=K(0..7),   v[8..15]=K(16..23)  of col N=lane
//   lane>=16: v[0..7]=K(8..15),  v[8..15]=K(24..31)  of col N=lane-16
__global__ void k_pack_b(const float* __restrict__ W, _Float16* __restrict__ dst,
                         int Ntiles, int Ktiles, int ld) {
  int tid = blockIdx.x * blockDim.x + threadIdx.x;
  if (tid >= Ntiles * Ktiles * 32) return;
  int lane = tid & 31, tile = tid >> 5;
  int kt = tile % Ktiles, nt = tile / Ktiles;
  int c = lane & 15, hi = lane >> 4;
  int kb = kt * 32 + hi * 8;
  v16h v;
#pragma unroll
  for (int i = 0; i < 16; ++i)
    v[i] = (_Float16)W[(size_t)(kb + i + (i >= 8 ? 8 : 0)) * ld + nt * 16 + c];
  *(v16h*)(dst + (size_t)tid * 16) = v;
}

// Each wave computes a 16x64 strip: one shared A fragment x 4 B fragments per
// K-step -> 4 v_wmma_f32_16x16x32_f16 per 5 contiguous-32B fragment loads
// (vs 1 wmma per 2 loads for a single tile). Ntiles must be divisible by 4.
__global__ void k_gemm_wmma(const _Float16* __restrict__ Af, const _Float16* __restrict__ Bf,
                            float* __restrict__ C, int Mtiles, int Ntiles, int Ktiles, int ldc) {
  int wave = (blockIdx.x * blockDim.x + threadIdx.x) >> 5;
  int lane = threadIdx.x & 31;
  int Ngrp = Ntiles >> 2;
  if (wave >= Mtiles * Ngrp) return;
  int ng = wave % Ngrp, mt = wave / Ngrp;
  const v16h* A  = (const v16h*)Af + (size_t)mt * Ktiles * 32;
  const v16h* B0 = (const v16h*)Bf + (size_t)(ng * 4 + 0) * Ktiles * 32;
  const v16h* B1 = (const v16h*)Bf + (size_t)(ng * 4 + 1) * Ktiles * 32;
  const v16h* B2 = (const v16h*)Bf + (size_t)(ng * 4 + 2) * Ktiles * 32;
  const v16h* B3 = (const v16h*)Bf + (size_t)(ng * 4 + 3) * Ktiles * 32;
  v8f acc0 = {}, acc1 = {}, acc2 = {}, acc3 = {};
#pragma unroll 2
  for (int kt = 0; kt < Ktiles; ++kt) {
    int idx = kt * 32 + lane;
    v16h a  = A[idx];
    v16h b0 = B0[idx];
    v16h b1 = B1[idx];
    v16h b2 = B2[idx];
    v16h b3 = B3[idx];
    acc0 = __builtin_amdgcn_wmma_f32_16x16x32_f16(false, a, false, b0, (short)0, acc0, false, false);
    acc1 = __builtin_amdgcn_wmma_f32_16x16x32_f16(false, a, false, b1, (short)0, acc1, false, false);
    acc2 = __builtin_amdgcn_wmma_f32_16x16x32_f16(false, a, false, b2, (short)0, acc2, false, false);
    acc3 = __builtin_amdgcn_wmma_f32_16x16x32_f16(false, a, false, b3, (short)0, acc3, false, false);
  }
  // D layout: lane<16 -> rows 0..7, lane>=16 -> rows 8..15; col = lane%16
  int col0 = ng * 64 + (lane & 15);
  int rbase = mt * 16 + (lane >> 4) * 8;
#pragma unroll
  for (int i = 0; i < 8; ++i) {
    float* rowp = C + (size_t)(rbase + i) * ldc + col0;
    rowp[0]  = acc0[i];
    rowp[16] = acc1[i];
    rowp[32] = acc2[i];
    rowp[48] = acc3[i];
  }
}

// alpha_src[n,h] = dot(h[n,h,:], att_src[h,:]); same for dst. Thread per (n,h).
__global__ void k_node_alpha(const float* __restrict__ feat, const float* __restrict__ aS,
                             const float* __restrict__ aD, float* __restrict__ oS,
                             float* __restrict__ oD, int nnodes, int H, int C) {
  int tid = blockIdx.x * blockDim.x + threadIdx.x;
  if (tid >= nnodes * H) return;
  int h = tid % H, n = tid / H;
  const float* f = feat + (size_t)n * H * C + (size_t)h * C;
  const float* s = aS + (size_t)h * C;
  const float* d = aD + (size_t)h * C;
  float accS = 0.f, accD = 0.f;
  for (int c = 0; c < C; ++c) { float v = f[c]; accS += v * s[c]; accD += v * d[c]; }
  oS[tid] = accS;
  oD[tid] = accD;
}

// leaky_relu(alpha_src[src]+alpha_dst[dst]); store per-edge; segment max via atomicMax.
__global__ void k_edge_alpha_max(const int* __restrict__ ei, const float* __restrict__ aS,
                                 const float* __restrict__ aD, float* __restrict__ alpha,
                                 float* __restrict__ amax, int H) {
  int tid = blockIdx.x * blockDim.x + threadIdx.x;
  if (tid >= EA * H) return;
  int h = tid % H, e = tid / H;
  int s = (e < NE) ? ei[e] : (e - NE);
  int d = (e < NE) ? ei[NE + e] : (e - NE);
  float a = aS[s * H + h] + aD[d * H + h];
  a = a > 0.f ? a : NEG * a;
  alpha[tid] = a;
  atomicMax(&amax[d * H + h], a);
}

// ex = exp(alpha - amax[dst]); stored in place; denom[dst] += ex.
__global__ void k_edge_exp_sum(const int* __restrict__ ei, float* __restrict__ alpha,
                               const float* __restrict__ amax, float* __restrict__ den, int H) {
  int tid = blockIdx.x * blockDim.x + threadIdx.x;
  if (tid >= EA * H) return;
  int h = tid % H, e = tid / H;
  int d = (e < NE) ? ei[NE + e] : (e - NE);
  float ex = expf(alpha[tid] - amax[d * H + h]);
  alpha[tid] = ex;
  atomicAdd(&den[d * H + h], ex);
}

// One wave per (edge, head): agg[dst,h,:] += feat[src,h,:] * coef. Coalesced 32-lane passes.
__global__ void k_edge_aggregate(const int* __restrict__ ei, const float* __restrict__ ex,
                                 const float* __restrict__ den, const float* __restrict__ feat,
                                 float* __restrict__ agg, int H, int C) {
  int gw = (blockIdx.x * blockDim.x + threadIdx.x) >> 5;
  int lane = threadIdx.x & 31;
  if (gw >= EA * H) return;
  int h = gw % H, e = gw / H;
  int s = (e < NE) ? ei[e] : (e - NE);
  int d = (e < NE) ? ei[NE + e] : (e - NE);
  float coef = ex[gw] / (den[d * H + h] + 1e-16f);
  const float* f = feat + (size_t)s * H * C + (size_t)h * C;
  float* o = agg + (size_t)d * H * C + (size_t)h * C;
  int R = C >> 5;
  for (int r = 0; r < R; ++r) {
    int c = lane + (r << 5);
    atomicAdd(o + c, f[c] * coef);
  }
}

// One wave per node: out = log_softmax(agg + b2). C=256 -> 8 values/lane.
__global__ void k_bias_logsoftmax(const float* __restrict__ agg, const float* __restrict__ b,
                                  float* __restrict__ out) {
  int gw = (blockIdx.x * blockDim.x + threadIdx.x) >> 5;
  int lane = threadIdx.x & 31;
  if (gw >= NN) return;
  const float* row = agg + (size_t)gw * OUTC;
  float v[8];
  float mx = -3.0e38f;
#pragma unroll
  for (int r = 0; r < 8; ++r) { int c = lane + 32 * r; v[r] = row[c] + b[c]; mx = fmaxf(mx, v[r]); }
  for (int o = 16; o > 0; o >>= 1) mx = fmaxf(mx, __shfl_xor(mx, o, 32));
  float sum = 0.f;
#pragma unroll
  for (int r = 0; r < 8; ++r) sum += expf(v[r] - mx);
  for (int o = 16; o > 0; o >>= 1) sum += __shfl_xor(sum, o, 32);
  float l = mx + logf(sum);
#pragma unroll
  for (int r = 0; r < 8; ++r) out[(size_t)gw * OUTC + lane + 32 * r] = v[r] - l;
}

// ---------------- host-side orchestration ----------------
static inline int cdiv(long a, long b) { return (int)((a + b - 1) / b); }

extern "C" void kernel_launch(void* const* d_in, const int* in_sizes, int n_in,
                              void* d_out, int out_size, void* d_ws, size_t ws_size,
                              hipStream_t stream) {
  (void)in_sizes; (void)n_in; (void)out_size; (void)ws_size;
  const float* x   = (const float*)d_in[0];
  const int*   ei  = (const int*)  d_in[1];
  const float* W1  = (const float*)d_in[2];
  const float* as1 = (const float*)d_in[3];
  const float* ad1 = (const float*)d_in[4];
  const float* b1  = (const float*)d_in[5];
  const float* W2  = (const float*)d_in[6];
  const float* as2 = (const float*)d_in[7];
  const float* ad2 = (const float*)d_in[8];
  const float* b2  = (const float*)d_in[9];
  float* out = (float*)d_out;
  char*  ws  = (char*)d_ws;

  _Float16* xfrag = (_Float16*)(ws + OFF_XFRAG);   // layer1 A-frags; reused for layer2 (h2)
  _Float16* w1f   = (_Float16*)(ws + OFF_W1F);
  _Float16* w2f   = (_Float16*)(ws + OFF_W2F);
  float* h1   = (float*)(ws + OFF_H1);             // [NN,512]; reused as g [NN,256]
  float* nAs  = (float*)(ws + OFF_AS);
  float* nAd  = (float*)(ws + OFF_AD);
  float* amax = (float*)(ws + OFF_AMAX);
  float* den  = (float*)(ws + OFF_DEN);
  float* ebuf = (float*)(ws + OFF_EDGE);
  float* agg  = (float*)(ws + OFF_AGG);            // [NN,512]; reused as agg2 [NN,256]

  const dim3 B(256);

  // ---- pack inputs to f16 WMMA fragments ----
  k_pack_a<<<cdiv((long)MT * KT1 * 32, 256), B, 0, stream>>>(x, xfrag, MT, KT1, INC);
  k_pack_b<<<cdiv((long)NT1 * KT1 * 32, 256), B, 0, stream>>>(W1, w1f, NT1, KT1, HC1);
  k_pack_b<<<cdiv((long)NT2 * KT2 * 32, 256), B, 0, stream>>>(W2, w2f, NT2, KT2, OUTC);

  // ---- layer 1 ----
  k_gemm_wmma<<<cdiv((long)MT * (NT1 / 4), 8), B, 0, stream>>>(xfrag, w1f, h1, MT, NT1, KT1, HC1);
  k_node_alpha<<<cdiv((long)NN * H1H, 256), B, 0, stream>>>(h1, as1, ad1, nAs, nAd, NN, H1H, HID);
  k_fill<<<cdiv((long)NN * H1H, 256), B, 0, stream>>>(amax, -1.0e30f, NN * H1H);
  k_fill<<<cdiv((long)NN * H1H, 256), B, 0, stream>>>(den, 0.f, NN * H1H);
  k_fill<<<cdiv((long)NN * HC1, 256), B, 0, stream>>>(agg, 0.f, NN * HC1);
  k_edge_alpha_max<<<cdiv((long)EA * H1H, 256), B, 0, stream>>>(ei, nAs, nAd, ebuf, amax, H1H);
  k_edge_exp_sum<<<cdiv((long)EA * H1H, 256), B, 0, stream>>>(ei, ebuf, amax, den, H1H);
  k_edge_aggregate<<<cdiv((long)EA * H1H, 8), B, 0, stream>>>(ei, ebuf, den, h1, agg, H1H, HID);

  // ---- layer1 epilogue: (+b1, ELU) packed directly into layer2 A-frags ----
  k_pack_a_elu_bias<<<cdiv((long)MT * KT2 * 32, 256), B, 0, stream>>>(agg, b1, xfrag, MT, KT2, HC1);

  // ---- layer 2 (g overwrites h1 region; agg2 overwrites agg region) ----
  k_gemm_wmma<<<cdiv((long)MT * (NT2 / 4), 8), B, 0, stream>>>(xfrag, w2f, h1, MT, NT2, KT2, OUTC);
  k_node_alpha<<<cdiv((long)NN, 256), B, 0, stream>>>(h1, as2, ad2, nAs, nAd, NN, 1, OUTC);
  k_fill<<<cdiv((long)NN, 256), B, 0, stream>>>(amax, -1.0e30f, NN);
  k_fill<<<cdiv((long)NN, 256), B, 0, stream>>>(den, 0.f, NN);
  k_fill<<<cdiv((long)NN * OUTC, 256), B, 0, stream>>>(agg, 0.f, NN * OUTC);
  k_edge_alpha_max<<<cdiv((long)EA, 256), B, 0, stream>>>(ei, nAs, nAd, ebuf, amax, 1);
  k_edge_exp_sum<<<cdiv((long)EA, 256), B, 0, stream>>>(ei, ebuf, amax, den, 1);
  k_edge_aggregate<<<cdiv((long)EA, 8), B, 0, stream>>>(ei, ebuf, den, h1, agg, 1, OUTC);

  // ---- bias + log_softmax ----
  k_bias_logsoftmax<<<cdiv((long)NN, 8), B, 0, stream>>>(agg, b2, out);
}
